// NeRFOriginal_13400297964169
// MI455X (gfx1250) — matrix operand order repaired and checked by reading, not compiled, not run on hardware
//
#include <hip/hip_runtime.h>

// ---------------------------------------------------------------------------
// NeRF MLP fused forward for MI455X (gfx1250, wave32, WMMA).
// Compute-bound (~311 GFLOP vs ~100MB traffic) -> f16 WMMA 16x16x32, fp32 acc.
// Pack pass converts fp32 weights to f16 WMMA B-fragment layout in d_ws.
// Main kernel: 1 block = 64 points, 8 waves; activations ping-pong in LDS f16.
// Input tile staged via GLOBAL_LOAD_ASYNC_TO_LDS_B128 (ASYNCcnt) DMA path.
// ---------------------------------------------------------------------------

#define ACT_STRIDE 328   // halves per LDS activation row (>=320, 16B aligned, bank-spread)

typedef __attribute__((ext_vector_type(16))) _Float16 v16h;
typedef __attribute__((ext_vector_type(8)))  _Float16 v8h;
typedef __attribute__((ext_vector_type(8)))  float    v8f;

struct NerfParams {
  const float* x;
  const _Float16* w[12];   // packed f16 fragments: w0..w7, feat, alpha, views, rgb
  const float* b[12];      // fp32 biases (same order)
  float* out;
  int npts;
};

// ---- CDNA5 async LDS DMA (inline asm; no builtin coverage) ----------------
// GVS mode: global addr = SADDR(64) + VADDR(32-bit byte offset); per-lane 16B.
__device__ __forceinline__ void async_load_b128(unsigned lds_byte_addr,
                                                unsigned gbyte_off,
                                                const void* sbase) {
  asm volatile("global_load_async_to_lds_b128 %0, %1, %2"
               :: "v"(lds_byte_addr), "v"(gbyte_off), "s"(sbase)
               : "memory");
}
__device__ __forceinline__ void wait_asynccnt0() {
  asm volatile("s_wait_asynccnt 0x0" ::: "memory");
}
// Low 32 bits of a generic LDS pointer == logical LDS byte address (ISA 10.2).
__device__ __forceinline__ unsigned lds_addr32(const void* p) {
  return (unsigned)(unsigned long long)p;
}

// ---- fragment loads -------------------------------------------------------
// A fragment (16x32 f16), ISA layout: lanes 0-15 row m=lane, K = k0+0..7 (h0-7),
// k0+16..23 (h8-15); lanes 16-31 row m=lane-16, K offsets +8.
__device__ __forceinline__ v16h load_a_frag(const _Float16* act, int stride, int lane, int k0) {
  int m   = lane & 15;
  int sel = (lane >> 4) & 1;
  const _Float16* base = act + m * stride + k0 + sel * 8;
  v8h lo = *(const v8h*)(base);        // ds_load_b128
  v8h hi = *(const v8h*)(base + 16);   // ds_load_b128
  v16h a;
#pragma unroll
  for (int i = 0; i < 8; ++i) { a[i] = lo[i]; a[8 + i] = hi[i]; }
  return a;
}

// B fragment (32x16 f16) pre-packed: one contiguous 32B per lane.
__device__ __forceinline__ v16h load_b_frag(const _Float16* wp, int ksteps, int ntile,
                                            int kstep, int lane) {
  return *(const v16h*)(wp + ((size_t)((ntile * ksteps + kstep) * 32 + lane)) * 16);
}

__device__ __forceinline__ v8f wmma_f16(v16h a, v16h b, v8f c) {
  return __builtin_amdgcn_wmma_f32_16x16x32_f16(false, a, false, b, (short)0, c, false, false);
}

// ---- one MLP layer: 64 rows (4 M-tiles), NT n-tiles per wave --------------
template<int KSTEPS, int NT, bool RELU>
__device__ __forceinline__ void layer_mlp(const _Float16* __restrict__ actIn, int strideIn,
                                          _Float16* __restrict__ actOut, int strideOut,
                                          const _Float16* __restrict__ wp,
                                          const float* __restrict__ bias, int nact,
                                          int wv, int lane) {
  const v8f vzero = {};
  v8f acc[4][NT];
#pragma unroll
  for (int mt = 0; mt < 4; ++mt)
#pragma unroll
    for (int j = 0; j < NT; ++j) acc[mt][j] = vzero;

#pragma unroll
  for (int ks = 0; ks < KSTEPS; ++ks) {
    if (ks + 1 < KSTEPS)   // warm next weight fragment in near caches
      __builtin_prefetch(wp + ((size_t)((wv * KSTEPS + ks + 1) * 32 + lane)) * 16, 0, 3);
    v16h b[NT];
#pragma unroll
    for (int j = 0; j < NT; ++j) b[j] = load_b_frag(wp, KSTEPS, wv + 8 * j, ks, lane);
#pragma unroll
    for (int mt = 0; mt < 4; ++mt) {
      v16h a = load_a_frag(actIn + mt * 16 * strideIn, strideIn, lane, ks * 32);
#pragma unroll
      for (int j = 0; j < NT; ++j) acc[mt][j] = wmma_f16(a, b[j], acc[mt][j]);
    }
  }

  // Epilogue: D layout lanes 0-15 -> n=lane, m=r; lanes 16-31 -> n=lane-16, m=r+8.
  int mb = (lane >> 4) << 3;
#pragma unroll
  for (int j = 0; j < NT; ++j) {
    int n = (wv + 8 * j) * 16 + (lane & 15);
    float bv = (n < nact) ? bias[n] : 0.0f;
#pragma unroll
    for (int mt = 0; mt < 4; ++mt)
#pragma unroll
      for (int r = 0; r < 8; ++r) {
        float v = acc[mt][j][r] + bv;
        if (RELU) v = fmaxf(v, 0.0f);
        actOut[(mt * 16 + mb + r) * strideOut + n] = (_Float16)v;
      }
  }
}

// ---- narrow head (alpha / rgb): one padded n-tile, 4 M-tiles --------------
template<int KSTEPS>
__device__ __forceinline__ void head_acc(const _Float16* __restrict__ actIn, int strideIn,
                                         const _Float16* __restrict__ wp, int lane,
                                         v8f acc[4]) {
  const v8f vzero = {};
#pragma unroll
  for (int mt = 0; mt < 4; ++mt) acc[mt] = vzero;
#pragma unroll
  for (int ks = 0; ks < KSTEPS; ++ks) {
    v16h b = load_b_frag(wp, KSTEPS, 0, ks, lane);
#pragma unroll
    for (int mt = 0; mt < 4; ++mt) {
      v16h a = load_a_frag(actIn + mt * 16 * strideIn, strideIn, lane, ks * 32);
      acc[mt] = wmma_f16(a, b, acc[mt]);
    }
  }
}

// ---- main fused kernel ----------------------------------------------------
__global__ __launch_bounds__(256, 2) void nerf_fused(NerfParams p) {
  __shared__ alignas(16) float    xStage[64 * 90];        // async DMA landing (fp32)
  __shared__ alignas(16) _Float16 posS[64 * 64];          // pos padded 63->64
  __shared__ alignas(16) _Float16 vwsS[64 * 32];          // views padded 27->32
  __shared__ alignas(16) _Float16 bufA[64 * ACT_STRIDE];
  __shared__ alignas(16) _Float16 bufB[64 * ACT_STRIDE];

  int tid  = threadIdx.x;
  int lane = tid & 31;
  int wv   = tid >> 5;
  int row0 = blockIdx.x * 64;

  // ---- async DMA: 64 rows x 90 fp32 = 23040 B = 1440 x 16B, 16B-aligned ----
  {
    const float* gbase = p.x + (size_t)row0 * 90;
    unsigned lds0 = lds_addr32(xStage);
    for (int c = tid; c < 1440; c += 256)
      async_load_b128(lds0 + (unsigned)c * 16u, (unsigned)c * 16u, gbase);
    wait_asynccnt0();
  }
  __syncthreads();

  // convert staged fp32 -> f16 pos/views buffers
  for (int idx = tid; idx < 64 * 90; idx += 256) {
    int m = idx / 90, c = idx % 90;
    float v = xStage[idx];
    if (c < 63) posS[m * 64 + c] = (_Float16)v;
    else        vwsS[m * 32 + (c - 63)] = (_Float16)v;
  }
  if (tid < 64) posS[tid * 64 + 63] = (_Float16)0.0f;
  for (int idx = tid; idx < 64 * 5; idx += 256) {
    int m = idx / 5, c = idx % 5;
    vwsS[m * 32 + 27 + c] = (_Float16)0.0f;
  }
  __syncthreads();

  // layer 0: K=64
  layer_mlp<2, 2, true>(posS, 64, bufA, ACT_STRIDE, p.w[0], p.b[0], 256, wv, lane);
  __syncthreads();
  layer_mlp<8, 2, true>(bufA, ACT_STRIDE, bufB, ACT_STRIDE, p.w[1], p.b[1], 256, wv, lane);
  __syncthreads();
  layer_mlp<8, 2, true>(bufB, ACT_STRIDE, bufA, ACT_STRIDE, p.w[2], p.b[2], 256, wv, lane);
  __syncthreads();
  layer_mlp<8, 2, true>(bufA, ACT_STRIDE, bufB, ACT_STRIDE, p.w[3], p.b[3], 256, wv, lane);
  __syncthreads();
  // layer 4: write y into cols 64..319 of bufA; pos goes to cols 0..63 (skip concat)
  layer_mlp<8, 2, true>(bufB, ACT_STRIDE, bufA + 64, ACT_STRIDE, p.w[4], p.b[4], 256, wv, lane);
  for (int idx = tid; idx < 64 * 64; idx += 256) {
    int m = idx >> 6, c = idx & 63;
    bufA[m * ACT_STRIDE + c] = posS[idx];
  }
  __syncthreads();
  // layer 5: K=320 (63 pos + pad + 256 y; weight rows remapped in pack pass)
  layer_mlp<10, 2, true>(bufA, ACT_STRIDE, bufB, ACT_STRIDE, p.w[5], p.b[5], 256, wv, lane);
  __syncthreads();
  layer_mlp<8, 2, true>(bufB, ACT_STRIDE, bufA, ACT_STRIDE, p.w[6], p.b[6], 256, wv, lane);
  __syncthreads();
  layer_mlp<8, 2, true>(bufA, ACT_STRIDE, bufB, ACT_STRIDE, p.w[7], p.b[7], 256, wv, lane);
  __syncthreads();

  // alpha head (wave 0) reads final y in bufB; no relu
  if (wv == 0) {
    v8f acc[4];
    head_acc<8>(bufB, ACT_STRIDE, p.w[9], lane, acc);
    if ((lane & 15) == 0) {
      float ab = p.b[9][0];
      int mb = (lane >> 4) << 3;
#pragma unroll
      for (int mt = 0; mt < 4; ++mt)
#pragma unroll
        for (int r = 0; r < 8; ++r)
          p.out[(size_t)(row0 + mt * 16 + mb + r) * 4 + 3] = acc[mt][r] + ab;
    }
  }
  // feature (no relu): bufB -> bufA cols 0..255; views appended at 256..287
  layer_mlp<8, 2, false>(bufB, ACT_STRIDE, bufA, ACT_STRIDE, p.w[8], p.b[8], 256, wv, lane);
  for (int idx = tid; idx < 64 * 32; idx += 256) {
    int m = idx >> 5, c = idx & 31;
    bufA[m * ACT_STRIDE + 256 + c] = vwsS[idx];
  }
  __syncthreads();
  // views layer: K=288, N=128 (1 n-tile per wave), relu
  layer_mlp<9, 1, true>(bufA, ACT_STRIDE, bufB, ACT_STRIDE, p.w[10], p.b[10], 128, wv, lane);
  __syncthreads();
  // rgb head (wave 0): K=128, no relu
  if (wv == 0) {
    v8f acc[4];
    head_acc<4>(bufB, ACT_STRIDE, p.w[11], lane, acc);
    int n = lane & 15;
    if (n < 3) {
      float rb = p.b[11][n];
      int mb = (lane >> 4) << 3;
#pragma unroll
      for (int mt = 0; mt < 4; ++mt)
#pragma unroll
        for (int r = 0; r < 8; ++r)
          p.out[(size_t)(row0 + mt * 16 + mb + r) * 4 + n] = acc[mt][r] + rb;
    }
  }
  // second reference output: zeros (npts x 3)
  size_t zbase = (size_t)p.npts * 4;
  for (int idx = tid; idx < 64 * 3; idx += 256) {
    int m = idx / 3, c = idx % 3;
    p.out[zbase + (size_t)(row0 + m) * 3 + c] = 0.0f;
  }
}

// ---- weight pack: fp32 (K,N) row-major -> f16 WMMA B-fragments ------------
// dst element t: h=t&15, lane=(t>>4)&31, frag=t>>9; frag = ntile*ksteps+kstep.
// Element = W[k][n] with n = ntile*16 + lane%16, k = kstep*32 + (h/8)*16 + (lane/16)*8 + h%8.
// skip_at>=0: padded row `skip_at` is zero, rows above map to src row k-1
// (used for the layer-5 skip concat [pos(63) | pad | y(256)]).
__global__ void pack_weight(const float* __restrict__ src, _Float16* __restrict__ dst,
                            int K, int N, int ksteps, int ntiles, int skip_at) {
  int t = blockIdx.x * blockDim.x + threadIdx.x;
  int total = ntiles * ksteps * 512;
  if (t >= total) return;
  int h     = t & 15;
  int lane  = (t >> 4) & 31;
  int f     = t >> 9;
  int kstep = f % ksteps;
  int ntile = f / ksteps;
  int n  = ntile * 16 + (lane & 15);
  int kp = kstep * 32 + ((h >> 3) << 4) + ((lane >> 4) << 3) + (h & 7);
  int k  = kp;
  bool ok = true;
  if (skip_at >= 0) {
    if (kp == skip_at)      ok = false;
    else if (kp > skip_at)  k = kp - 1;
  }
  float v = 0.0f;
  if (ok && k < K && n < N) v = src[(size_t)k * N + n];
  dst[t] = (_Float16)v;
}

// ---------------------------------------------------------------------------
extern "C" void kernel_launch(void* const* d_in, const int* in_sizes, int n_in,
                              void* d_out, int out_size, void* d_ws, size_t ws_size,
                              hipStream_t stream) {
  NerfParams p;
  p.x = (const float*)d_in[0];
  const float* wsrc[12];
  for (int i = 0; i < 8; ++i) {
    wsrc[i] = (const float*)d_in[1 + 2 * i];
    p.b[i]  = (const float*)d_in[2 + 2 * i];
  }
  wsrc[8]  = (const float*)d_in[17]; p.b[8]  = (const float*)d_in[18];  // feat
  wsrc[9]  = (const float*)d_in[19]; p.b[9]  = (const float*)d_in[20];  // alpha
  wsrc[10] = (const float*)d_in[21]; p.b[10] = (const float*)d_in[22];  // views
  wsrc[11] = (const float*)d_in[23]; p.b[11] = (const float*)d_in[24];  // rgb
  p.out  = (float*)d_out;
  int npts = in_sizes[0] / 90;
  p.npts = npts;

  static const int Ks[12]  = {63,256,256,256,256,319,256,256, 256,256,283,128};
  static const int Kps[12] = {64,256,256,256,256,320,256,256, 256,256,288,128};
  static const int Ns[12]  = {256,256,256,256,256,256,256,256, 256,  1,128,  3};
  static const int Nps[12] = {256,256,256,256,256,256,256,256, 256, 16,128, 16};
  static const int Sk[12]  = { -1, -1, -1, -1, -1, 63, -1, -1,  -1, -1, -1, -1};

  _Float16* base = (_Float16*)d_ws;
  size_t off = 0;
  for (int i = 0; i < 12; ++i) {
    int ksteps = Kps[i] / 32;
    int ntiles = Nps[i] / 16;
    int total  = ksteps * ntiles * 512;
    p.w[i] = base + off;
    pack_weight<<<(total + 255) / 256, 256, 0, stream>>>(
        wsrc[i], base + off, Ks[i], Ns[i], ksteps, ntiles, Sk[i]);
    off += (size_t)total;
  }

  nerf_fused<<<npts / 64, 256, 0, stream>>>(p);
}
